// MultiHeadFeedForward_23398981829315
// MI455X (gfx1250) — matrix operand order repaired and checked by reading, not compiled, not run on hardware
//
#include <hip/hip_runtime.h>

// MultiHeadFeedForward (per-head SwiGLU MLP) for MI455X / gfx1250.
// Round 4: same branch-free pipeline as round 3; async weight staging now uses
// a single asm block per 32KB chunk with INST_OFFSET immediates (ISA adds the
// offset to both the global and LDS addresses), eliminating per-copy v_add_u64
// address math. Weights pre-converted to f16 [head][chunk][N][K] by a prep
// kernel; compute is v_wmma_f32_16x16x32_f16 with f32 accumulation; x/out are
// non-temporal streams so L2 keeps the reused f16 weights.

typedef __attribute__((ext_vector_type(16))) _Float16 v16h;
typedef __attribute__((ext_vector_type(8)))  _Float16 v8h;
typedef __attribute__((ext_vector_type(4)))  _Float16 v4h;
typedef __attribute__((ext_vector_type(8)))  float    v8f;
typedef __attribute__((ext_vector_type(4)))  float    v4f;

#define EMB       2048
#define HEADS     16
#define HEAD_DIM  128
#define FFN       512
#define MTILE     64          // tokens per workgroup
#define NCH       128         // N (or K) chunk staged in LDS at a time
#define NTOK      16384       // 4 * 4096
#define WELEMS    (HEADS * 4 * NCH * NCH)   // f16 elems per converted tensor

// ---- CDNA5 async global->LDS copy (ISA ch.10 async ops, ASYNCcnt) ----------
// One contiguous 32KB f16 chunk (128x128): 256 threads x 8 x b128.
// INST_OFFSET is applied to BOTH the LDS and global addresses (ISA 15.18.3),
// so one LDS-offset VGPR + one 64-bit address pair covers all 8 transfers.
__device__ __forceinline__ void prefetch_chunk(_Float16* dst, const _Float16* src,
                                               int tid) {
  unsigned lds  = (unsigned)(uintptr_t)dst + (unsigned)tid * 16u;
  const char* g = (const char*)src + (size_t)tid * 16;
  asm volatile(
      "global_load_async_to_lds_b128 %0, %1, off\n\t"
      "global_load_async_to_lds_b128 %0, %1, off offset:4096\n\t"
      "global_load_async_to_lds_b128 %0, %1, off offset:8192\n\t"
      "global_load_async_to_lds_b128 %0, %1, off offset:12288\n\t"
      "global_load_async_to_lds_b128 %0, %1, off offset:16384\n\t"
      "global_load_async_to_lds_b128 %0, %1, off offset:20480\n\t"
      "global_load_async_to_lds_b128 %0, %1, off offset:24576\n\t"
      "global_load_async_to_lds_b128 %0, %1, off offset:28672"
      :: "v"(lds), "v"(g) : "memory");
}
__device__ __forceinline__ void wait_async8() {
  asm volatile("s_wait_asynccnt 0x8" ::: "memory");
}
__device__ __forceinline__ void wait_async0() {
  asm volatile("s_wait_asynccnt 0x0" ::: "memory");
}

// ---- WMMA fragment gathers (ISA 7.12.2 wave32 layouts) ----------------------
__device__ __forceinline__ v16h frag_A(const _Float16* base, int row, int kbase,
                                       int stride, int lane) {
  const _Float16* p = base + (size_t)(row + (lane & 15)) * stride
                           + kbase + ((lane >> 4) << 3);
  v8h lo = *(const v8h*)(p);
  v8h hi = *(const v8h*)(p + 16);
  return __builtin_shufflevector(lo, hi, 0,1,2,3,4,5,6,7,8,9,10,11,12,13,14,15);
}
__device__ __forceinline__ v16h frag_B(const _Float16* base, int col, int kbase,
                                       int stride, int lane) {
  const _Float16* p = base + (size_t)(col + (lane & 15)) * stride
                           + kbase + ((lane >> 4) << 3);
  v8h lo = *(const v8h*)(p);
  v8h hi = *(const v8h*)(p + 16);
  return __builtin_shufflevector(lo, hi, 0,1,2,3,4,5,6,7,8,9,10,11,12,13,14,15);
}

// ---- prep: f32 weights -> f16, transposed [head][chunk][n][k] ---------------
__global__ void prep_weights_kernel(const float* __restrict__ gw,
                                    const float* __restrict__ uw,
                                    const float* __restrict__ dw,
                                    _Float16* __restrict__ gt,
                                    _Float16* __restrict__ ut,
                                    _Float16* __restrict__ dt) {
  int idx = blockIdx.x * 256 + threadIdx.x;      // over 16*4*128*128 = 1M elems
  int n = idx & 127;
  int k = (idx >> 7) & 127;
  int c = (idx >> 14) & 3;
  int h = idx >> 16;
  int t = blockIdx.y;                             // 0=gate, 1=up, 2=down
  const float* src = (t == 0) ? gw : ((t == 1) ? uw : dw);
  _Float16*    dst = (t == 0) ? gt : ((t == 1) ? ut : dt);
  size_t soff = (t == 2)
      ? ((size_t)h * 65536 + (size_t)(c * 128 + k) * 128 + n)   // down: [512][128]
      : ((size_t)h * 65536 + (size_t)k * 512 + c * 128 + n);    // gate/up: [128][512]
  dst[(size_t)h * 65536 + (size_t)c * 16384 + (size_t)n * 128 + k] =
      (_Float16)src[soff];
}

// ---- main fused kernel ------------------------------------------------------
__global__ __launch_bounds__(256, 2)
void MultiHeadFeedForward_23398981829315_kernel(
    const float* __restrict__ x, float* __restrict__ out,
    const _Float16* __restrict__ gt, const _Float16* __restrict__ ut,
    const _Float16* __restrict__ dt) {
  __shared__ _Float16 sA[MTILE * HEAD_DIM];   // x tile f16, row-major  (16 KB)
  __shared__ _Float16 sH[MTILE * FFN];        // SwiGLU output         (64 KB)
  __shared__ _Float16 sW[2][NCH * NCH];       // weight double-buffer  (64 KB)

  const int tid  = threadIdx.x;
  const int lane = tid & 31;
  const int wid  = tid >> 5;
  const int m0   = (wid >> 1) << 4;
  const int nh   = (wid & 1) * 64;

  const int head = blockIdx.y;
  const int t0   = blockIdx.x * MTILE;

  const _Float16* gh = gt + (size_t)head * 4 * (NCH * NCH);
  const _Float16* uh = ut + (size_t)head * 4 * (NCH * NCH);
  const _Float16* dh = dt + (size_t)head * 4 * (NCH * NCH);

  // start gate chunk 0 flowing while we stage the x tile
  prefetch_chunk(sW[0], gh, tid);

  // x tile (64 x 128) -> f16 LDS, row-major; non-temporal (single-use stream)
  const float* xs = x + (size_t)t0 * EMB + head * HEAD_DIM;
  for (int i = tid; i < (MTILE * HEAD_DIM) / 4; i += 256) {
    int r  = i >> 5;
    int c4 = (i & 31) << 2;
    v4f v = __builtin_nontemporal_load((const v4f*)(xs + (size_t)r * EMB + c4));
    v4h h;
    h[0] = (_Float16)v[0]; h[1] = (_Float16)v[1];
    h[2] = (_Float16)v[2]; h[3] = (_Float16)v[3];
    *(v4h*)(&sA[r * HEAD_DIM + c4]) = h;
  }
  __syncthreads();

  v16h axf[4];
#pragma unroll
  for (int ks = 0; ks < 4; ++ks) axf[ks] = frag_A(sA, m0, ks * 32, HEAD_DIM, lane);

  const v8f vzero = {0.f, 0.f, 0.f, 0.f, 0.f, 0.f, 0.f, 0.f};
  const int rb = m0 + ((lane >> 4) << 3);

  // ===== stage 1: gate -> sW[0], up -> sW[1], branch-free pipeline ==========
  for (int c = 0; c < 4; ++c) {
    // ---- gate chunk c (already in flight into sW[0]) ----
    prefetch_chunk(sW[1], uh + (size_t)c * (NCH * NCH), tid);   // up_c -> sW[1]
    wait_async8();                 // gate_c landed; up_c in flight
    __syncthreads();
    v8f accg[4];
#pragma unroll
    for (int t = 0; t < 4; ++t) accg[t] = vzero;
#pragma unroll
    for (int ks = 0; ks < 4; ++ks) {
#pragma unroll
      for (int t = 0; t < 4; ++t) {
        v16h b = frag_B(sW[0], nh + t * 16, ks * 32, NCH, lane);
        accg[t] = __builtin_amdgcn_wmma_f32_16x16x32_f16(
            false, axf[ks], false, b, (short)0, accg[t], false, false);
      }
    }
    __syncthreads();               // everyone done reading sW[0]

    // ---- up chunk c ----
    const _Float16* nxt = (c < 3) ? (gh + (size_t)(c + 1) * (NCH * NCH)) : dh;
    prefetch_chunk(sW[0], nxt, tid);                            // next -> sW[0]
    wait_async8();                 // up_c landed; next in flight
    __syncthreads();
    v8f accu[4];
#pragma unroll
    for (int t = 0; t < 4; ++t) accu[t] = vzero;
#pragma unroll
    for (int ks = 0; ks < 4; ++ks) {
#pragma unroll
      for (int t = 0; t < 4; ++t) {
        v16h b = frag_B(sW[1], nh + t * 16, ks * 32, NCH, lane);
        accu[t] = __builtin_amdgcn_wmma_f32_16x16x32_f16(
            false, axf[ks], false, b, (short)0, accu[t], false, false);
      }
    }
    // SwiGLU on matching C/D layouts; scatter f16 into sH
#pragma unroll
    for (int t = 0; t < 4; ++t) {
      int col = c * NCH + nh + t * 16 + (lane & 15);
#pragma unroll
      for (int i = 0; i < 8; ++i) {
        float g = accg[t][i];
        float s = g / (1.0f + __expf(-g));  // silu
        sH[(rb + i) * FFN + col] = (_Float16)(s * accu[t][i]);
      }
    }
    __syncthreads();               // everyone done reading sW[1]
  }

  // ===== stage 2: out = h @ Wd, alternating buffers, d0 already in sW[0] ====
  v8f acco[4];
#pragma unroll
  for (int t = 0; t < 4; ++t) acco[t] = vzero;

#pragma unroll
  for (int kc = 0; kc < 4; ++kc) {
    if (kc < 3) {
      prefetch_chunk(sW[(kc + 1) & 1], dh + (size_t)(kc + 1) * (NCH * NCH), tid);
      wait_async8();
    } else {
      wait_async0();
    }
    __syncthreads();
    const _Float16* wb = sW[kc & 1];
#pragma unroll
    for (int ks = 0; ks < 4; ++ks) {
      v16h a = frag_A(sH, m0, kc * NCH + ks * 32, FFN, lane);
#pragma unroll
      for (int t = 0; t < 4; ++t) {
        v16h b = frag_B(wb, nh + t * 16, ks * 32, NCH, lane);
        acco[t] = __builtin_amdgcn_wmma_f32_16x16x32_f16(
            false, a, false, b, (short)0, acco[t], false, false);
      }
    }
    __syncthreads();
  }

  // fp32 output store, non-temporal (single-use stream)
  float* os = out + (size_t)t0 * EMB + head * HEAD_DIM;
#pragma unroll
  for (int t = 0; t < 4; ++t) {
    int col = nh + t * 16 + (lane & 15);
#pragma unroll
    for (int i = 0; i < 8; ++i)
      __builtin_nontemporal_store(acco[t][i], os + (size_t)(rb + i) * EMB + col);
  }
}

extern "C" void kernel_launch(void* const* d_in, const int* in_sizes, int n_in,
                              void* d_out, int out_size, void* d_ws, size_t ws_size,
                              hipStream_t stream) {
  const float* x  = (const float*)d_in[0];
  const float* gw = (const float*)d_in[1];
  const float* uw = (const float*)d_in[2];
  const float* dw = (const float*)d_in[3];
  float* out = (float*)d_out;

  _Float16* gt = (_Float16*)d_ws;              // 2 MB each, 6 MB total
  _Float16* ut = gt + WELEMS;
  _Float16* dt = ut + WELEMS;

  dim3 pgrid(WELEMS / 256, 3);                 // 4096 x 3 blocks
  prep_weights_kernel<<<pgrid, 256, 0, stream>>>(gw, uw, dw, gt, ut, dt);

  dim3 grid(NTOK / MTILE, HEADS);              // 256 token tiles x 16 heads
  MultiHeadFeedForward_23398981829315_kernel<<<grid, 256, 0, stream>>>(
      x, out, gt, ut, dt);

  (void)in_sizes; (void)n_in; (void)out_size; (void)ws_size;
}
